// DynamicNodeAttention_90606630076482
// MI455X (gfx1250) — matrix operand order repaired and checked by reading, not compiled, not run on hardware
//
#include <hip/hip_runtime.h>
#include <hip/hip_bf16.h>
#include <math.h>

// ---------------------------------------------------------------------------
// DynamicNodeAttention for MI455X (gfx1250, wave32, WMMA 16x16x32 f16)
//   b=2, n=4096, HIDDEN=256, HEADS=8, HEAD_DIM=32
// ---------------------------------------------------------------------------

#define HIDDEN 256
#define HEADS  8
#define HDIM   32
#define NSEQ   4096
#define NB     2
#define ROWS   (NB * NSEQ)   // 8192

typedef __attribute__((ext_vector_type(16))) _Float16 v16h;
typedef __attribute__((ext_vector_type(8)))  float    v8f;
typedef __attribute__((ext_vector_type(4)))  float    f32x4;
typedef __attribute__((ext_vector_type(4)))  unsigned int u32x4;

union HFrag {
    v16h v;
    _Float16 h[16];
    unsigned int u[8];
};

// Raw v_exp_f32 (base-2 exponential on AMD hardware)
#define EXP2F(x) __builtin_amdgcn_exp2f(x)

// 1/sqrt(32) * log2(e): folds both the score scale and the exp2 base change
// into the Q projection.
#define QSCALE 0.25501822405905606f

// ---------------------------------------------------------------------------
// Kernel 1: q/k/v = node_state @ W{q,k,v}.T, cast to f16.
//   q pre-scaled by QSCALE; v stored transposed [b,h,d,n].
//   One 16x16 output tile per wave, K accumulated in steps of 32.
// ---------------------------------------------------------------------------
__global__ __launch_bounds__(256) void qkv_proj(
    const float* __restrict__ X,
    const float* __restrict__ Wq, const float* __restrict__ Wk,
    const float* __restrict__ Wv,
    _Float16* __restrict__ qo, _Float16* __restrict__ ko,
    _Float16* __restrict__ vo)
{
    const int lane = threadIdx.x & 31;
    const int wave = threadIdx.x >> 5;
    const int hs   = lane >> 4;     // half-wave select
    const int ln   = lane & 15;
    const int tile = blockIdx.x * 8 + wave;       // 8192 tiles: 512 x 16
    const int rt   = tile >> 4;
    const int ct   = tile & 15;
    const int z    = blockIdx.z;                  // 0:q 1:k 2:v
    const float* W = (z == 0) ? Wq : (z == 1 ? Wk : Wv);

    const int arow = rt * 16 + ln;
    const int ncol = ct * 16 + ln;
    v8f acc = {};
    for (int kk = 0; kk < HIDDEN; kk += 32) {
        HFrag A, B;
        // A fragment (16x32 f16): lane<16 holds K {0..7,16..23}, lane>=16 +8
        const float* xp = X + (size_t)arow * HIDDEN + kk + hs * 8;
#pragma unroll
        for (int r = 0; r < 4; ++r) {
            A.h[2*r]   = (_Float16)xp[2*r];
            A.h[2*r+1] = (_Float16)xp[2*r+1];
            A.h[2*r+8] = (_Float16)xp[16 + 2*r];
            A.h[2*r+9] = (_Float16)xp[16 + 2*r+1];
        }
        // B fragment = W.T chunk (K x N): element (k, n) = W[n][k]
        const float* wp = W + (size_t)ncol * HIDDEN + kk + hs * 8;
#pragma unroll
        for (int r = 0; r < 4; ++r) {
            B.h[2*r]   = (_Float16)wp[2*r];
            B.h[2*r+1] = (_Float16)wp[2*r+1];
            B.h[2*r+8] = (_Float16)wp[16 + 2*r];
            B.h[2*r+9] = (_Float16)wp[16 + 2*r+1];
        }
        acc = __builtin_amdgcn_wmma_f32_16x16x32_f16(
            false, A.v, false, B.v, (short)0, acc, false, false);
    }
    if (z == 2) {
        // V transposed [b,h,d,n]: lane's 8 rows are consecutive n -> one b128
        union { _Float16 hh[8]; u32x4 q4; } pk;
#pragma unroll
        for (int i = 0; i < 8; ++i) pk.hh[i] = (_Float16)acc[i];
        int grow0 = rt * 16 + 8 * hs;            // first row of this half
        int gcol  = ct * 16 + ln;
        int b  = grow0 >> 12, nr0 = grow0 & (NSEQ - 1);
        int h  = gcol >> 5,   d   = gcol & 31;
        *(u32x4*)&vo[(((size_t)b * HEADS + h) * HDIM + d) * NSEQ + nr0] = pk.q4;
    } else {
        const float scale = (z == 0) ? QSCALE : 1.0f;
        _Float16* dst = (z == 0) ? qo : ko;
#pragma unroll
        for (int i = 0; i < 8; ++i) {
            int grow = rt * 16 + i + 8 * hs;     // C layout: M = i + 8*hs
            int gcol = ct * 16 + ln;             //           N = lane & 15
            int b  = grow >> 12, nr = grow & (NSEQ - 1);
            int h  = gcol >> 5,  d  = gcol & 31;
            dst[(((size_t)b * HEADS + h) * NSEQ + nr) * HDIM + d] =
                (_Float16)(acc[i] * scale);
        }
    }
}

// ---------------------------------------------------------------------------
// Kernel 2: flash attention with log-adjacency bias (base-2 domain).
//   Block = 256 threads = 8 waves = 8 heads, one 16-row q tile, k-step 64.
//   Adjacency tile: software-pipelined nontemporal loads, log2'ed once,
//   shared across heads via LDS.
// ---------------------------------------------------------------------------
__global__ __launch_bounds__(256) void flash_attn(
    const _Float16* __restrict__ q, const _Float16* __restrict__ k,
    const _Float16* __restrict__ vT, const float* __restrict__ adj,
    float* __restrict__ att)
{
    __shared__ float    ldsBias[16 * 64];        // log2(adj+1e-6) tile (4 KB)
    __shared__ _Float16 ldsP[8][16 * 64];        // per-wave P slabs   (16 KB)

    const int lane  = threadIdx.x & 31;
    const int wave  = threadIdx.x >> 5;          // = head
    const int hs    = lane >> 4;
    const int ln    = lane & 15;
    const int b     = blockIdx.y;
    const int qbase = blockIdx.x * 16;
    const int bh    = b * HEADS + wave;

    const unsigned int* qu = (const unsigned int*)(q  + (size_t)bh * NSEQ * HDIM);
    const unsigned int* ku = (const unsigned int*)(k  + (size_t)bh * NSEQ * HDIM);
    const unsigned int* vu = (const unsigned int*)(vT + (size_t)bh * HDIM * NSEQ);
    const float* adjq = adj + ((size_t)b * NSEQ + qbase) * NSEQ;

    // Q fragment (A layout, 16 rows x 32 dims; HEAD_DIM == WMMA K)
    HFrag Qa;
    {
        const unsigned int* p = qu + (size_t)(qbase + ln) * 16 + hs * 4;
#pragma unroll
        for (int r = 0; r < 4; ++r) {
            Qa.u[r]     = p[r];       // dims {hs*8 + 0..7}
            Qa.u[r + 4] = p[8 + r];   // dims {hs*8 + 16..23}
        }
    }

    // software-pipelined adjacency: this thread owns 4 floats per tile
    const int prow = threadIdx.x >> 4;           // 0..15
    const int pcol = (threadIdx.x & 15) * 4;     // 0..60
    f32x4 pre = __builtin_nontemporal_load(
        (const f32x4*)(adjq + (size_t)prow * NSEQ + pcol));

    float rmax[8], rsum[8];
    v8f Oa = {}, Ob = {};
#pragma unroll
    for (int i = 0; i < 8; ++i) { rmax[i] = -3.0e38f; rsum[i] = 0.f; }

    for (int kb = 0; kb < NSEQ; kb += 64) {
        // --- publish bias tile for kb, prefetch kb+64 ---
        __syncthreads();
        {
            float* bp = ldsBias + prow * 64 + pcol;
            bp[0] = __log2f(pre.x + 1e-6f);
            bp[1] = __log2f(pre.y + 1e-6f);
            bp[2] = __log2f(pre.z + 1e-6f);
            bp[3] = __log2f(pre.w + 1e-6f);
        }
        if (kb + 64 < NSEQ) {
            pre = __builtin_nontemporal_load(
                (const f32x4*)(adjq + (size_t)prow * NSEQ + kb + 64 + pcol));
        }
        __syncthreads();

        // --- S = Q @ K^T over 64 k-columns (scale+log2e folded into Q) ---
        HFrag Bk[4];
#pragma unroll
        for (int j = 0; j < 4; ++j)
#pragma unroll
            for (int r = 0; r < 8; ++r)
                Bk[j].u[r] = ku[(size_t)(kb + 16*j + ln) * 16 + hs * 4 + (r & 3) + (r >> 2) * 8];
        v8f cz = {};
        v8f S[4];
#pragma unroll
        for (int j = 0; j < 4; ++j)
            S[j] = __builtin_amdgcn_wmma_f32_16x16x32_f16(
                false, Qa.v, false, Bk[j].v, (short)0, cz, false, false);

        // --- bias add + online softmax (base 2), rows in 16-lane groups ---
#pragma unroll
        for (int i = 0; i < 8; ++i) {
            int M = i + 8 * hs;
            const float* brow = ldsBias + M * 64 + ln;
            S[0][i] += brow[0];
            S[1][i] += brow[16];
            S[2][i] += brow[32];
            S[3][i] += brow[48];
            float m = fmaxf(fmaxf(S[0][i], S[1][i]), fmaxf(S[2][i], S[3][i]));
            m = fmaxf(m, __shfl_xor(m, 1));
            m = fmaxf(m, __shfl_xor(m, 2));
            m = fmaxf(m, __shfl_xor(m, 4));
            m = fmaxf(m, __shfl_xor(m, 8));
            float mn = fmaxf(rmax[i], m);
            float al = EXP2F(rmax[i] - mn);
            float p0 = EXP2F(S[0][i] - mn);
            float p1 = EXP2F(S[1][i] - mn);
            float p2 = EXP2F(S[2][i] - mn);
            float p3 = EXP2F(S[3][i] - mn);
            float ts = (p0 + p1) + (p2 + p3);
            ts += __shfl_xor(ts, 1);
            ts += __shfl_xor(ts, 2);
            ts += __shfl_xor(ts, 4);
            ts += __shfl_xor(ts, 8);
            rsum[i] = rsum[i] * al + ts;
            rmax[i] = mn;
            Oa[i] *= al;
            Ob[i] *= al;
            S[0][i] = p0; S[1][i] = p1; S[2][i] = p2; S[3][i] = p3;
        }

        // --- P: C layout -> f16 A layout via this wave's private LDS slab ---
        _Float16* lp = ldsP[wave];
#pragma unroll
        for (int i = 0; i < 8; ++i) {
            int M = i + 8 * hs;
            lp[M * 64 + ln]      = (_Float16)S[0][i];
            lp[M * 64 + 16 + ln] = (_Float16)S[1][i];
            lp[M * 64 + 32 + ln] = (_Float16)S[2][i];
            lp[M * 64 + 48 + ln] = (_Float16)S[3][i];
        }
        HFrag Pa0, Pa1;
        {
            const unsigned int* lu = (const unsigned int*)lp;
#pragma unroll
            for (int r = 0; r < 4; ++r) {
                Pa0.u[r]     = lu[ln * 32 + hs * 4 + r];
                Pa0.u[r + 4] = lu[ln * 32 + hs * 4 + 8 + r];
                Pa1.u[r]     = lu[ln * 32 + 16 + hs * 4 + r];
                Pa1.u[r + 4] = lu[ln * 32 + 16 + hs * 4 + 8 + r];
            }
        }

        // --- O += P @ V (V pre-transposed -> contiguous K-pair loads) ---
        HFrag Bv0a, Bv0b, Bv1a, Bv1b;
#pragma unroll
        for (int r = 0; r < 8; ++r) {
            int kp0 = (kb >> 1) + hs * 4 + (r & 3) + (r >> 2) * 8;            // k-chunk 0
            int kp1 = ((kb + 32) >> 1) + hs * 4 + (r & 3) + (r >> 2) * 8;     // k-chunk 1
            Bv0a.u[r] = vu[(size_t)ln        * (NSEQ / 2) + kp0];
            Bv0b.u[r] = vu[(size_t)(16 + ln) * (NSEQ / 2) + kp0];
            Bv1a.u[r] = vu[(size_t)ln        * (NSEQ / 2) + kp1];
            Bv1b.u[r] = vu[(size_t)(16 + ln) * (NSEQ / 2) + kp1];
        }
        Oa = __builtin_amdgcn_wmma_f32_16x16x32_f16(
            false, Pa0.v, false, Bv0a.v, (short)0, Oa, false, false);
        Ob = __builtin_amdgcn_wmma_f32_16x16x32_f16(
            false, Pa0.v, false, Bv0b.v, (short)0, Ob, false, false);
        Oa = __builtin_amdgcn_wmma_f32_16x16x32_f16(
            false, Pa1.v, false, Bv1a.v, (short)0, Oa, false, false);
        Ob = __builtin_amdgcn_wmma_f32_16x16x32_f16(
            false, Pa1.v, false, Bv1b.v, (short)0, Ob, false, false);
    }

    // --- epilogue: normalize and merge heads into [b, n, 256] fp32 ---
#pragma unroll
    for (int i = 0; i < 8; ++i) {
        int M = i + 8 * hs;
        float inv = 1.0f / rsum[i];
        size_t rowg = (size_t)b * NSEQ + qbase + M;
        att[rowg * HIDDEN + wave * HDIM + ln]      = Oa[i] * inv;
        att[rowg * HIDDEN + wave * HDIM + 16 + ln] = Ob[i] * inv;
    }
}

// ---------------------------------------------------------------------------
// Kernel 3: out = att @ Wo.T + bo  (fp32 output)
// ---------------------------------------------------------------------------
__global__ __launch_bounds__(256) void out_proj(
    const float* __restrict__ att, const float* __restrict__ Wo,
    const float* __restrict__ bo, float* __restrict__ out)
{
    const int lane = threadIdx.x & 31;
    const int wave = threadIdx.x >> 5;
    const int hs   = lane >> 4;
    const int ln   = lane & 15;
    const int tile = blockIdx.x * 8 + wave;
    const int rt   = tile >> 4;
    const int ct   = tile & 15;

    const int arow = rt * 16 + ln;
    const int ncol = ct * 16 + ln;
    v8f acc = {};
    for (int kk = 0; kk < HIDDEN; kk += 32) {
        HFrag A, B;
        const float* xp = att + (size_t)arow * HIDDEN + kk + hs * 8;
#pragma unroll
        for (int r = 0; r < 4; ++r) {
            A.h[2*r]   = (_Float16)xp[2*r];
            A.h[2*r+1] = (_Float16)xp[2*r+1];
            A.h[2*r+8] = (_Float16)xp[16 + 2*r];
            A.h[2*r+9] = (_Float16)xp[16 + 2*r+1];
        }
        const float* wp = Wo + (size_t)ncol * HIDDEN + kk + hs * 8;
#pragma unroll
        for (int r = 0; r < 4; ++r) {
            B.h[2*r]   = (_Float16)wp[2*r];
            B.h[2*r+1] = (_Float16)wp[2*r+1];
            B.h[2*r+8] = (_Float16)wp[16 + 2*r];
            B.h[2*r+9] = (_Float16)wp[16 + 2*r+1];
        }
        acc = __builtin_amdgcn_wmma_f32_16x16x32_f16(
            false, A.v, false, B.v, (short)0, acc, false, false);
    }
#pragma unroll
    for (int i = 0; i < 8; ++i) {
        int grow = rt * 16 + i + 8 * hs;
        int gcol = ct * 16 + ln;
        out[(size_t)grow * HIDDEN + gcol] = acc[i] + bo[gcol];
    }
}

// ---------------------------------------------------------------------------
extern "C" void kernel_launch(void* const* d_in, const int* in_sizes, int n_in,
                              void* d_out, int out_size, void* d_ws, size_t ws_size,
                              hipStream_t stream)
{
    const float* node = (const float*)d_in[0];  // (2, 4096, 256)
    const float* adj  = (const float*)d_in[1];  // (2, 4096, 4096)
    const float* Wq   = (const float*)d_in[2];
    const float* Wk   = (const float*)d_in[3];
    const float* Wv   = (const float*)d_in[4];
    const float* Wo   = (const float*)d_in[5];
    const float* bo   = (const float*)d_in[6];
    float* out = (float*)d_out;                 // (2, 4096, 256) fp32

    char* ws = (char*)d_ws;
    _Float16* qf  = (_Float16*)(ws);                        // 4 MB
    _Float16* kf  = (_Float16*)(ws + ((size_t)4  << 20));   // 4 MB
    _Float16* vT  = (_Float16*)(ws + ((size_t)8  << 20));   // 4 MB
    float*    att = (float*)   (ws + ((size_t)12 << 20));   // 8 MB

    dim3 blk(256);
    qkv_proj  <<<dim3(1024, 1, 3),    blk, 0, stream>>>(node, Wq, Wk, Wv, qf, kf, vT);
    flash_attn<<<dim3(NSEQ / 16, NB), blk, 0, stream>>>(qf, kf, vT, adj, att);
    out_proj  <<<dim3(1024),          blk, 0, stream>>>(att, Wo, bo, out);
}
